// VectorQuantizer_61297773248861
// MI455X (gfx1250) — matrix operand order repaired and checked by reading, not compiled, not run on hardware
//
#include <hip/hip_runtime.h>

// ---------------------------------------------------------------------------
// VQ-VAE quantizer for MI455X (gfx1250, wave32, WMMA + TDM).
// Distance GEMM as bf16x3 (hi/lo split) on v_wmma_f32_16x16x32_bf16.
// Codebook tiles DMA'd into LDS by the Tensor Data Mover, double-buffered.
// ---------------------------------------------------------------------------

typedef __attribute__((ext_vector_type(16))) __bf16        v16bf;
typedef __attribute__((ext_vector_type(8)))  float         v8f;
typedef __attribute__((ext_vector_type(4)))  unsigned int  v4u;
typedef __attribute__((ext_vector_type(8)))  unsigned int  v8u;

union ABReg { v16bf v; uint4 q[2]; };

__device__ __forceinline__ unsigned short bf16_rne(float f) {
  unsigned int u = __float_as_uint(f);
  unsigned int r = u + 0x7FFFu + ((u >> 16) & 1u);   // round-to-nearest-even
  return (unsigned short)(r >> 16);
}
__device__ __forceinline__ float bf16_f32(unsigned short h) {
  return __uint_as_float(((unsigned int)h) << 16);
}

__device__ __forceinline__ v8f wmma_bf16(const ABReg& a, const ABReg& b, v8f c) {
  return __builtin_amdgcn_wmma_f32_16x16x32_bf16(false, a.v, false, b.v,
                                                 (short)0, c, false, false);
}

// 2D TDM: copy a 16-row x 128-col bf16 tile (row stride 128 elems in memory)
// into LDS at lds_off, padding each 256B row by 16B (-> 272B rows in LDS).
// D# group0: count=1, lds_addr, global_addr, type=2.
// D# group1: data_size=2B, pad_enable, pad_interval=64DW, pad_amount=4DW,
//            tensor_dim0=128, tensor_dim1=1024, tile_dim0=128, tile_dim1=16,
//            tensor_dim0_stride=128.
__device__ __forceinline__ void tdm_load_tile(unsigned lds_off,
                                              unsigned long long gaddr) {
  v4u g0;
  g0[0] = 1u;                                   // count = 1 valid descriptor
  g0[1] = lds_off;                              // LDS byte address
  g0[2] = (unsigned)gaddr;                      // global_addr[31:0]
  g0[3] = (unsigned)(gaddr >> 32) | (2u << 30); // global_addr[56:32] | type=2
  v8u g1;
  g1[0] = (1u << 16)        // data_size = 1 -> 2 bytes
        | (1u << 20)        // pad_enable
        | (5u << 22)        // pad_interval: 64 DWORDs (one 256B row)
        | (3u << 25);       // pad_amount: 4 DWORDs (16B)
  g1[1] = (128u << 16);     // tensor_dim0 = 128 (lo16)
  g1[2] = (1024u << 16);    // tensor_dim0 hi = 0 | tensor_dim1 = 1024 (lo16)
  g1[3] = (128u << 16);     // tensor_dim1 hi = 0 | tile_dim0 = 128
  g1[4] = 16u;              // tile_dim1 = 16, tile_dim2 = 0
  g1[5] = 128u;             // tensor_dim0_stride = 128 (lo32)
  g1[6] = 0u;
  g1[7] = 0u;
  asm volatile("tensor_load_to_lds %0, %1" :: "s"(g0), "s"(g1) : "memory");
}

// Problem constants: B=16, C=128, H=64, W=64, K=1024.
#define NROWS   65536      // B*H*W
#define CDIM    128
#define KCODES  1024
#define NELEM   8388608    // B*C*H*W
#define LDSS    136        // 128 + 8 bf16 pad -> conflict-free ds_load_b128
#define BT_EL   (16 * LDSS)

// ---------------------------------------------------------------------------
// Kernel 1: codebook -> bf16 hi/lo planes, half-norms, zero usage flags.
// ---------------------------------------------------------------------------
__global__ __launch_bounds__(128) void vq_prep(
    const float* __restrict__ cb,
    unsigned short* __restrict__ cbh, unsigned short* __restrict__ cbl,
    float* __restrict__ hn, int* __restrict__ flags) {
  const int k = blockIdx.x;
  const int c = threadIdx.x;
  float v = cb[k * CDIM + c];
  unsigned short hi = bf16_rne(v);
  cbh[k * CDIM + c] = hi;
  cbl[k * CDIM + c] = bf16_rne(v - bf16_f32(hi));
  float s = v * v;
  #pragma unroll
  for (int off = 16; off >= 1; off >>= 1) s += __shfl_xor(s, off, 32);
  __shared__ float red[4];
  if ((c & 31) == 0) red[c >> 5] = s;
  __syncthreads();
  if (c == 0) {
    hn[k] = 0.5f * (red[0] + red[1] + red[2] + red[3]);
    flags[k] = 0;
  }
}

// ---------------------------------------------------------------------------
// Kernel 2: WMMA distance GEMM + argmin.
// 128 threads (4 waves), 128 rows per block, 32 rows per wave (2 C-tiles
// sharing every B fragment). B tiles (16 codes x 128, hi+lo) are TDM-loaded
// into double-buffered LDS; wave 0 prefetches tile k+1 while all compute k.
// ---------------------------------------------------------------------------
__global__ __launch_bounds__(128) void vq_argmin(
    const float* __restrict__ z,
    const unsigned short* __restrict__ cbh,
    const unsigned short* __restrict__ cbl,
    const float* __restrict__ hn,
    int* __restrict__ idxs, int* __restrict__ flags) {
  __shared__ __align__(16) unsigned short zs_hi[128 * LDSS];
  __shared__ __align__(16) unsigned short zs_lo[128 * LDSS];
  __shared__ __align__(16) unsigned short btile[2][2][BT_EL];  // [buf][hi/lo]

  const int rowbase = blockIdx.x << 7;            // 128 rows / block
  const int b = rowbase >> 12;                    // n = b*4096 + h*64 + w
  const size_t base0 = (size_t)b * 524288 + (size_t)(rowbase & 4095);

  // ---- stage z tile (128 rows x 128 c) as bf16 hi/lo; coalesced 512B runs.
  #pragma unroll 4
  for (int c = 0; c < CDIM; ++c) {
    float v = z[base0 + (size_t)c * 4096 + threadIdx.x];
    unsigned short hi = bf16_rne(v);
    zs_hi[threadIdx.x * LDSS + c] = hi;
    zs_lo[threadIdx.x * LDSS + c] = bf16_rne(v - bf16_f32(hi));
  }

  const int wave   = threadIdx.x >> 5;
  const int lane   = threadIdx.x & 31;
  const int lane15 = lane & 15;
  const int sel    = lane >> 4;

  const unsigned long long gah = (unsigned long long)(uintptr_t)cbh;
  const unsigned long long gal = (unsigned long long)(uintptr_t)cbl;

  // ---- prologue: wave 0 kicks TDM for k-tile 0 into buffer 0.
  if (wave == 0) {
    tdm_load_tile((unsigned)(uintptr_t)&btile[0][0][0], gah);
    tdm_load_tile((unsigned)(uintptr_t)&btile[0][1][0], gal);
  }
  __syncthreads();                                // z staged (B not yet needed)

  // ---- hoist A fragments: 2 tiles x (hi,lo) x 4 k-chunks.
  // CDNA5 16-bit A layout: elems 0..7 -> K=c0+sel*8.., 8..15 -> K=c0+16+sel*8..
  ABReg a_hi[2][4], a_lo[2][4];
  #pragma unroll
  for (int t = 0; t < 2; ++t) {
    const int arow = (wave << 5) + (t << 4) + lane15;
    const unsigned short* zr_h = zs_hi + arow * LDSS;
    const unsigned short* zr_l = zs_lo + arow * LDSS;
    #pragma unroll
    for (int ci = 0; ci < 4; ++ci) {
      int c0 = (ci << 5) + (sel << 3);
      a_hi[t][ci].q[0] = *(const uint4*)(zr_h + c0);
      a_hi[t][ci].q[1] = *(const uint4*)(zr_h + c0 + 16);
      a_lo[t][ci].q[0] = *(const uint4*)(zr_l + c0);
      a_lo[t][ci].q[1] = *(const uint4*)(zr_l + c0 + 16);
    }
  }

  float mv[2][8];
  int   mi[2][8];
  #pragma unroll
  for (int t = 0; t < 2; ++t)
    #pragma unroll
    for (int r = 0; r < 8; ++r) { mv[t][r] = 3.4e38f; mi[t][r] = 0; }

  for (int kt = 0; kt < 64; ++kt) {
    const int buf = kt & 1;
    if (wave == 0) __builtin_amdgcn_s_wait_tensorcnt(0);  // tile kt landed
    __syncthreads();                                      // all may read buf
    if (wave == 0 && kt < 63) {
      // prefetch tile kt+1 into the other buffer (its readers finished at
      // iteration kt-1, ordered by the barrier above).
      unsigned long long goff = (unsigned long long)(kt + 1) * 4096ull;
      tdm_load_tile((unsigned)(uintptr_t)&btile[buf ^ 1][0][0], gah + goff);
      tdm_load_tile((unsigned)(uintptr_t)&btile[buf ^ 1][1][0], gal + goff);
    }

    const int krow = (kt << 4) + lane15;          // C-tile column = lane%16
    const unsigned short* bh_t = &btile[buf][0][0] + lane15 * LDSS;
    const unsigned short* bl_t = &btile[buf][1][0] + lane15 * LDSS;

    v8f acc0 = {}, acc1 = {};
    #pragma unroll
    for (int ci = 0; ci < 4; ++ci) {
      int c0 = (ci << 5) + (sel << 3);            // B: same K layout as A
      ABReg bh, bl;
      bh.q[0] = *(const uint4*)(bh_t + c0);
      bh.q[1] = *(const uint4*)(bh_t + c0 + 16);
      bl.q[0] = *(const uint4*)(bl_t + c0);
      bl.q[1] = *(const uint4*)(bl_t + c0 + 16);
      acc0 = wmma_bf16(a_hi[0][ci], bh, acc0);
      acc0 = wmma_bf16(a_hi[0][ci], bl, acc0);
      acc0 = wmma_bf16(a_lo[0][ci], bh, acc0);
      acc1 = wmma_bf16(a_hi[1][ci], bh, acc1);
      acc1 = wmma_bf16(a_hi[1][ci], bl, acc1);
      acc1 = wmma_bf16(a_lo[1][ci], bh, acc1);
    }
    // score = 0.5*||e||^2 - z.e  (same argmin ordering as full distance)
    float hnv = hn[krow];
    #pragma unroll
    for (int r = 0; r < 8; ++r) {
      float s0 = hnv - acc0[r];
      if (s0 < mv[0][r]) { mv[0][r] = s0; mi[0][r] = krow; }
      float s1 = hnv - acc1[r];
      if (s1 < mv[1][r]) { mv[1][r] = s1; mi[1][r] = krow; }
    }
  }

  // Min-reduce across the 16 lanes holding different columns (first-index ties).
  #pragma unroll
  for (int off = 8; off >= 1; off >>= 1) {
    #pragma unroll
    for (int t = 0; t < 2; ++t)
      #pragma unroll
      for (int r = 0; r < 8; ++r) {
        float ov = __shfl_xor(mv[t][r], off, 32);
        int   oi = __shfl_xor(mi[t][r], off, 32);
        if (ov < mv[t][r] || (ov == mv[t][r] && oi < mi[t][r])) {
          mv[t][r] = ov; mi[t][r] = oi;
        }
      }
  }

  if (lane15 == 0) {
    // lane 0 holds rows +0..7 (VGPR r -> row r), lane 16 rows +8..15
    #pragma unroll
    for (int t = 0; t < 2; ++t) {
      int rb = rowbase + (wave << 5) + (t << 4) + (sel << 3);
      #pragma unroll
      for (int r = 0; r < 8; ++r) {
        idxs[rb + r] = mi[t][r];
        flags[mi[t][r]] = 1;                      // benign race (all write 1)
      }
    }
  }
}

// ---------------------------------------------------------------------------
// Kernel 3: gather z_q (== z_q_st in value), write output, per-block
// squared-error partial sums (deterministic, no atomics).
// ---------------------------------------------------------------------------
__global__ __launch_bounds__(256) void vq_gather_loss(
    const float* __restrict__ z, const float* __restrict__ cb,
    const int* __restrict__ idxs,
    float* __restrict__ out, float* __restrict__ partials) {
  const int blk = blockIdx.x;                     // 32768 blocks
  const int bh  = blk >> 5;
  const int cg  = blk & 31;
  const int b = bh >> 6, h = bh & 63;
  const int w = threadIdx.x & 63;
  const int c = (cg << 2) + (threadIdx.x >> 6);
  const int n = (b << 12) + (h << 6) + w;
  const int k = idxs[n];
  float q  = cb[k * CDIM + c];
  size_t zi = (size_t)b * 524288 + (size_t)c * 4096 + (size_t)h * 64 + (size_t)w;
  float zv = z[zi];
  out[zi] = q;                                    // straight-through value
  float d = q - zv;
  float s = d * d;
  #pragma unroll
  for (int off = 16; off >= 1; off >>= 1) s += __shfl_xor(s, off, 32);
  __shared__ float red[8];
  if ((threadIdx.x & 31) == 0) red[threadIdx.x >> 5] = s;
  __syncthreads();
  if (threadIdx.x == 0) {
    float t = 0.f;
    #pragma unroll
    for (int i = 0; i < 8; ++i) t += red[i];
    partials[blk] = t;
  }
}

// ---------------------------------------------------------------------------
// Kernel 4: fixed-order reduction of partials + usage popcount -> scalars.
// ---------------------------------------------------------------------------
__global__ __launch_bounds__(256) void vq_finalize(
    const float* __restrict__ partials, const int* __restrict__ flags,
    float* __restrict__ out2) {
  __shared__ float fr[256];
  __shared__ int   ir[256];
  const int t = threadIdx.x;
  float s = 0.f;
  for (int i = t; i < 32768; i += 256) s += partials[i];
  int cnt = 0;
  for (int i = t; i < KCODES; i += 256) cnt += (flags[i] != 0);
  fr[t] = s; ir[t] = cnt;
  __syncthreads();
  for (int off = 128; off >= 1; off >>= 1) {
    if (t < off) { fr[t] += fr[t + off]; ir[t] += ir[t + off]; }
    __syncthreads();
  }
  if (t == 0) {
    out2[0] = 1.25f * fr[0] / (float)NELEM;       // codebook + 0.25*commitment
    out2[1] = (float)ir[0] / (float)KCODES;
  }
}

// ---------------------------------------------------------------------------
extern "C" void kernel_launch(void* const* d_in, const int* in_sizes, int n_in,
                              void* d_out, int out_size, void* d_ws, size_t ws_size,
                              hipStream_t stream) {
  const float* z  = (const float*)d_in[0];   // (16,128,64,64) f32
  const float* cb = (const float*)d_in[1];   // (1024,128) f32
  char* ws = (char*)d_ws;
  unsigned short* cbh   = (unsigned short*)(ws);             // 262144 B
  unsigned short* cbl   = (unsigned short*)(ws + 262144);    // 262144 B
  float*          hn    = (float*)(ws + 524288);             //   4096 B
  int*            idxs  = (int*)  (ws + 528384);             // 262144 B
  int*            flags = (int*)  (ws + 790528);             //   4096 B
  float*          parts = (float*)(ws + 794624);             // 131072 B
  float* out = (float*)d_out;                                // NELEM + 2 floats

  vq_prep       <<<dim3(KCODES),      dim3(128), 0, stream>>>(cb, cbh, cbl, hn, flags);
  vq_argmin     <<<dim3(NROWS / 128), dim3(128), 0, stream>>>(z, cbh, cbl, hn, idxs, flags);
  vq_gather_loss<<<dim3(32768),       dim3(256), 0, stream>>>(z, cb, idxs, out, parts);
  vq_finalize   <<<dim3(1),           dim3(256), 0, stream>>>(parts, flags, out + NELEM);
}
